// SchnetModel_64287070486528
// MI455X (gfx1250) — compile-verified, matrix operand order checked by reading, not run on hardware
//
#include <hip/hip_runtime.h>
#include <hip/hip_bf16.h>

typedef __attribute__((ext_vector_type(16))) _Float16 v16h;
typedef __attribute__((ext_vector_type(8)))  float    v8f;

#define NMAXC 512
#define EMAXC 16384
#define HC    128
#define LOG2F_C 0.6931471805599453f

// -------- helpers --------------------------------------------------------

// shifted softplus: softplus(x) - log(2), stable, using HW v_exp_f32/v_log_f32
__device__ __forceinline__ float sspf(float x) {
  float t = __expf(-fabsf(x));                 // v_exp_f32
  return fmaxf(x, 0.0f) + __logf(1.0f + t) - LOG2F_C;   // v_log_f32
}

// plain softplus (readout scale head)
__device__ __forceinline__ float softplusf(float x) {
  float t = __expf(-fabsf(x));
  return fmaxf(x, 0.0f) + __logf(1.0f + t);
}

// A-fragment (16xK tile, 16-bit) from LDS, per ISA 7.12.2 layout:
// lanes 0-15: M=lane, halves 0..7 = K 0..7, 8..15 = K 16..23 (of the 32-K tile)
// lanes 16-31: M=lane-16, halves 0..7 = K 8..15, 8..15 = K 24..31
__device__ __forceinline__ v16h load_afrag(const _Float16* X, int xstride, int kt, int lane) {
  const int m     = lane & 15;
  const int kbase = kt * 32 + ((lane & 16) ? 8 : 0);
  const _Float16* p = X + m * xstride + kbase;
  union { v16h v; uint4 q[2]; } u;
  u.q[0] = *(const uint4*)(p);        // K .. K+7   (ds_load_b128)
  u.q[1] = *(const uint4*)(p + 16);   // K+16..K+23 (ds_load_b128)
  return u.v;
}

// One 16x16 output tile of  X(16xK) @ W(KxN) + bias, K = nKt*32.
// Weights are pre-packed fragment-major: ((kt*NT+nt)*32 + lane)*16 halves.
__device__ __forceinline__ v8f gemm_tile(const _Float16* X, int xstride, int nKt,
                                         const _Float16* Wfrag, int NT, int nt,
                                         const float* bias, int lane) {
  const int n = nt * 16 + (lane & 15);
  const float bv = bias[n];
  v8f acc;
  #pragma unroll
  for (int r = 0; r < 8; ++r) acc[r] = bv;
  #pragma unroll
  for (int kt = 0; kt < nKt; ++kt) {
    v16h a = load_afrag(X, xstride, kt, lane);
    v16h w = *(const v16h*)(Wfrag + ((kt * NT + nt) * 32 + lane) * 16);  // 32B coalesced
    acc = __builtin_amdgcn_wmma_f32_16x16x32_f16(false, a, false, w, (short)0, acc,
                                                 false, false);
  }
  return acc;
}

// Store a C/D tile (f32) to an LDS 16x128 f16 tile, optional shifted-softplus.
// C/D layout: VGPR r -> M=r (lanes 0-15) / M=r+8 (lanes 16-31), N=lane%16.
__device__ __forceinline__ void store_act(v8f acc, _Float16* Y, int nt, int lane, bool act) {
  const int moff = (lane & 16) ? 8 : 0;
  const int n = nt * 16 + (lane & 15);
  #pragma unroll
  for (int r = 0; r < 8; ++r) {
    float v = acc[r];
    if (act) v = sspf(v);
    Y[(r + moff) * HC + n] = (_Float16)v;
  }
}

// -------- one-time prep kernels ------------------------------------------

// fp32 [K,N] row-major -> f16 fragment-major (zero padded to KT*32 x NT*16)
__global__ void convert_wfrag_kernel(const float* __restrict__ W, int K, int N,
                                     int KT, int NT, _Float16* __restrict__ dst) {
  int tid = blockIdx.x * blockDim.x + threadIdx.x;
  if (tid >= KT * NT * 512) return;
  int j    = tid & 15;
  int lane = (tid >> 4) & 31;
  int t    = tid >> 9;              // kt*NT + nt
  int nt   = t % NT;
  int kt   = t / NT;
  int n = nt * 16 + (lane & 15);
  int k = kt * 32 + ((j < 8) ? j : j + 8) + ((lane & 16) ? 8 : 0);
  float v = (k < K && n < N) ? W[k * N + n] : 0.0f;
  dst[tid] = (_Float16)v;
}

__global__ void embed_kernel(const int* __restrict__ nodes, const float* __restrict__ emb,
                             float* __restrict__ h, _Float16* __restrict__ h16) {
  int i = blockIdx.x * blockDim.x + threadIdx.x;   // over 32768*128
  int node = i >> 7, c = i & 127;
  float v = emb[nodes[node] * HC + c];
  h[i]   = v;
  h16[i] = (_Float16)v;
}

__global__ void fill0_kernel(float* __restrict__ p, int n) {
  int i = blockIdx.x * blockDim.x + threadIdx.x;
  if (i < n) p[i] = 0.0f;
}

// -------- fused per-edge interaction kernel ------------------------------
// 4 waves / block, each wave owns one 16-edge tile and its own LDS slice.
__global__ __launch_bounds__(128) void edge_msg_kernel(
    const int* __restrict__ edges, const float* __restrict__ edist,
    const int* __restrict__ num_edges, const _Float16* __restrict__ h16,
    const _Float16* __restrict__ wE1, const float* __restrict__ bE1,
    const _Float16* __restrict__ wE2, const float* __restrict__ bE2,
    const _Float16* __restrict__ wN1, const float* __restrict__ bN1,
    const _Float16* __restrict__ wN2, const float* __restrict__ bN2,
    float* __restrict__ msum) {
  __shared__ __align__(16) _Float16 sX[4][16 * HC];
  __shared__ __align__(16) _Float16 sY[4][16 * HC];
  __shared__ __align__(16) _Float16 sG[4][16 * HC];
  __shared__ int   sS[4][16];
  __shared__ int   sR[4][16];
  __shared__ float sD[4][16];

  const int wid  = threadIdx.x >> 5;
  const int lane = threadIdx.x & 31;
  const int tile = blockIdx.x * 4 + wid;          // 65536 tiles of 16 edges
  const int b     = tile >> 10;                   // EMAX/16 = 1024 tiles/graph
  const int eInG0 = (tile & 1023) * 16;
  const int e0    = tile * 16;
  const int ne    = num_edges[b];
  _Float16* X = sX[wid];
  _Float16* Y = sY[wid];
  _Float16* G = sG[wid];

  if (lane < 16) {
    int e = e0 + lane;
    sS[wid][lane] = edges[e * 2 + 0] + b * NMAXC;
    sR[wid][lane] = edges[e * 2 + 1] + b * NMAXC;
    sD[wid][lane] = edist[e];
  }
  // Gaussian expansion: 16x50 padded to 16x64, stride 64 in LDS
  #pragma unroll 4
  for (int t = 0; t < 32; ++t) {
    int id = t * 32 + lane;
    int m = id >> 6, j = id & 63;
    float dd = sD[wid][m] - 0.1f * (float)j;
    float g  = (j < 50) ? __expf(-dd * dd * 100.0f) : 0.0f;   // exp(-(d-mu)^2/step^2)
    X[m * 64 + j] = (_Float16)g;
  }
  // edge MLP: (16x64)@(64x128) -> ssp -> (16x128)@(128x128) = gates
  #pragma unroll
  for (int nt = 0; nt < 8; ++nt)
    store_act(gemm_tile(X, 64, 2, wE1, 8, nt, bE1, lane), Y, nt, lane, true);
  #pragma unroll
  for (int nt = 0; nt < 8; ++nt)
    store_act(gemm_tile(Y, HC, 4, wE2, 8, nt, bE2, lane), G, nt, lane, false);
  // gather sender node states (L2-resident h16) into X, stride 128
  #pragma unroll
  for (int t = 0; t < 8; ++t) {
    int id = t * 32 + lane;
    int m = id >> 4, c = (id & 15) * 8;
    *(uint4*)&X[m * HC + c] = *(const uint4*)(h16 + (size_t)sS[wid][m] * HC + c);
  }
  // node MLP on gathered states
  #pragma unroll
  for (int nt = 0; nt < 8; ++nt)
    store_act(gemm_tile(X, HC, 4, wN1, 8, nt, bN1, lane), Y, nt, lane, true);
  // second layer, gate, masked scatter-add
  const int moff = (lane & 16) ? 8 : 0;
  #pragma unroll
  for (int nt = 0; nt < 8; ++nt) {
    v8f acc = gemm_tile(Y, HC, 4, wN2, 8, nt, bN2, lane);
    const int n = nt * 16 + (lane & 15);
    #pragma unroll
    for (int r = 0; r < 8; ++r) {
      int m = r + moff;
      if (eInG0 + m < ne) {
        float msg = acc[r] * (float)G[m * HC + n];
        atomicAdd(&msum[(size_t)sR[wid][m] * HC + n], msg);
      }
    }
  }
}

// -------- per-node trans MLP + residual ----------------------------------
__global__ __launch_bounds__(128) void node_trans_kernel(
    const float* __restrict__ msum,
    const _Float16* __restrict__ wT1, const float* __restrict__ bT1,
    const _Float16* __restrict__ wT2, const float* __restrict__ bT2,
    float* __restrict__ h, _Float16* __restrict__ h16) {
  __shared__ __align__(16) _Float16 sX[4][16 * HC];
  __shared__ __align__(16) _Float16 sY[4][16 * HC];
  const int wid = threadIdx.x >> 5, lane = threadIdx.x & 31;
  const int tile = blockIdx.x * 4 + wid;       // 2048 tiles of 16 nodes
  const int n0 = tile * 16;
  _Float16* X = sX[wid];
  _Float16* Y = sY[wid];
  #pragma unroll 4
  for (int t = 0; t < 16; ++t) {
    int id = t * 32 + lane;                    // 0..511 float4s of the 16x128 tile
    float4 v = *(const float4*)(msum + (size_t)n0 * HC + id * 4);
    int base = id * 4;
    X[base + 0] = (_Float16)v.x;  X[base + 1] = (_Float16)v.y;
    X[base + 2] = (_Float16)v.z;  X[base + 3] = (_Float16)v.w;
  }
  #pragma unroll
  for (int nt = 0; nt < 8; ++nt)
    store_act(gemm_tile(X, HC, 4, wT1, 8, nt, bT1, lane), Y, nt, lane, true);
  const int moff = (lane & 16) ? 8 : 0;
  #pragma unroll
  for (int nt = 0; nt < 8; ++nt) {
    v8f acc = gemm_tile(Y, HC, 4, wT2, 8, nt, bT2, lane);
    const int n = nt * 16 + (lane & 15);
    #pragma unroll
    for (int r = 0; r < 8; ++r) {
      size_t idx = (size_t)(n0 + r + moff) * HC + n;
      float nv = h[idx] + acc[r];
      h[idx]   = nv;
      h16[idx] = (_Float16)nv;
    }
  }
}

// -------- readout ---------------------------------------------------------
__global__ __launch_bounds__(128) void readout_kernel(
    const _Float16* __restrict__ h16, const int* __restrict__ num_nodes,
    const _Float16* __restrict__ wR1, const float* __restrict__ bR1,
    const float* __restrict__ wR2, const float* __restrict__ bR2,
    float* __restrict__ gsum) {
  __shared__ __align__(16) _Float16 sX[4][16 * HC];
  __shared__ __align__(16) _Float16 sY[4][16 * HC];
  const int wid = threadIdx.x >> 5, lane = threadIdx.x & 31;
  const int tile = blockIdx.x * 4 + wid;
  const int n0 = tile * 16;
  _Float16* X = sX[wid];
  _Float16* Y = sY[wid];
  #pragma unroll
  for (int t = 0; t < 8; ++t) {
    int id = t * 32 + lane;
    int m = id >> 4, c = (id & 15) * 8;
    *(uint4*)&X[m * HC + c] = *(const uint4*)(h16 + (size_t)(n0 + m) * HC + c);
  }
  #pragma unroll
  for (int nt = 0; nt < 8; ++nt)
    store_act(gemm_tile(X, HC, 4, wR1, 8, nt, bR1, lane), Y, nt, lane, true);
  // final H->2 projection: lane = (c<<4)|m; 128-length dot from LDS
  const int c = lane >> 4, m = lane & 15;
  float dot = bR2[c];
  #pragma unroll 8
  for (int k = 0; k < HC; ++k) dot += (float)Y[m * HC + k] * wR2[k * 2 + c];
  const int b = n0 >> 9;                       // 512 nodes per graph
  if ((n0 & 511) + m < num_nodes[b]) atomicAdd(&gsum[b * 2 + c], dot);
}

__global__ void finalize_kernel(const float* __restrict__ gsum,
                                const int* __restrict__ num_nodes,
                                const float* __restrict__ loc_p,
                                const float* __restrict__ scale_p,
                                float* __restrict__ out) {
  int b = threadIdx.x;
  if (b < 64) {
    float loc = gsum[b * 2 + 0] * scale_p[0] + loc_p[0] * (float)num_nodes[b];
    float sc = softplusf(gsum[b * 2 + 1]) + 1e-6f;
    out[b]      = loc;
    out[64 + b] = sc;
  }
}

// -------- launcher --------------------------------------------------------
extern "C" void kernel_launch(void* const* d_in, const int* in_sizes, int n_in,
                              void* d_out, int out_size, void* d_ws, size_t ws_size,
                              hipStream_t stream) {
  (void)in_sizes; (void)n_in; (void)out_size; (void)ws_size;
  const int*   nodes     = (const int*)d_in[0];
  const int*   num_nodes = (const int*)d_in[1];
  const int*   edges     = (const int*)d_in[2];
  const float* edist     = (const float*)d_in[3];
  const int*   num_edges = (const int*)d_in[4];
  const float* emb       = (const float*)d_in[5];

  // workspace layout
  char* ws = (char*)d_ws;
  float*     h    = (float*)ws;                            // 32768*128 f32 = 16 MB
  _Float16*  h16  = (_Float16*)(ws + (size_t)16777216);    //  8 MB
  float*     msum = (float*)(ws + (size_t)25165824);       // 16 MB
  float*     gsum = (float*)(ws + (size_t)41943040);       // 512 B
  _Float16*  wf   = (_Float16*)(ws + (size_t)41943552);    // fragment weights

  const int FRAG_E1  = 2 * 8 * 512;   // 64x128
  const int FRAG_128 = 4 * 8 * 512;   // 128x128
  _Float16 *wE1[3], *wE2[3], *wN1[3], *wN2[3], *wT1[3], *wT2[3];
  size_t off = 0;
  for (int i = 0; i < 3; ++i) {
    wE1[i] = wf + off; off += FRAG_E1;
    wE2[i] = wf + off; off += FRAG_128;
    wN1[i] = wf + off; off += FRAG_128;
    wN2[i] = wf + off; off += FRAG_128;
    wT1[i] = wf + off; off += FRAG_128;
    wT2[i] = wf + off; off += FRAG_128;
  }
  _Float16* wR1 = wf + off;

  // weight conversion (fp32 -> fragment-major f16)
  for (int i = 0; i < 3; ++i) {
    int p = 6 + 12 * i;
    convert_wfrag_kernel<<<FRAG_E1 / 256, 256, 0, stream>>>((const float*)d_in[p + 0], 50, 128, 2, 8, wE1[i]);
    convert_wfrag_kernel<<<FRAG_128 / 256, 256, 0, stream>>>((const float*)d_in[p + 2], 128, 128, 4, 8, wE2[i]);
    convert_wfrag_kernel<<<FRAG_128 / 256, 256, 0, stream>>>((const float*)d_in[p + 4], 128, 128, 4, 8, wN1[i]);
    convert_wfrag_kernel<<<FRAG_128 / 256, 256, 0, stream>>>((const float*)d_in[p + 6], 128, 128, 4, 8, wN2[i]);
    convert_wfrag_kernel<<<FRAG_128 / 256, 256, 0, stream>>>((const float*)d_in[p + 8], 128, 128, 4, 8, wT1[i]);
    convert_wfrag_kernel<<<FRAG_128 / 256, 256, 0, stream>>>((const float*)d_in[p + 10], 128, 128, 4, 8, wT2[i]);
  }
  convert_wfrag_kernel<<<FRAG_128 / 256, 256, 0, stream>>>((const float*)d_in[42], 128, 128, 4, 8, wR1);

  // node embedding
  embed_kernel<<<(32768 * 128) / 256, 256, 0, stream>>>(nodes, emb, h, h16);

  // interactions
  for (int i = 0; i < 3; ++i) {
    int p = 6 + 12 * i;
    fill0_kernel<<<(32768 * 128) / 256, 256, 0, stream>>>(msum, 32768 * 128);
    edge_msg_kernel<<<(64 * EMAXC / 16) / 4, 128, 0, stream>>>(
        edges, edist, num_edges, h16,
        wE1[i], (const float*)d_in[p + 1], wE2[i], (const float*)d_in[p + 3],
        wN1[i], (const float*)d_in[p + 5], wN2[i], (const float*)d_in[p + 7], msum);
    node_trans_kernel<<<(32768 / 16) / 4, 128, 0, stream>>>(
        msum, wT1[i], (const float*)d_in[p + 9], wT2[i], (const float*)d_in[p + 11], h, h16);
  }

  // readout + normalize
  fill0_kernel<<<1, 128, 0, stream>>>(gsum, 128);
  readout_kernel<<<(32768 / 16) / 4, 128, 0, stream>>>(
      h16, num_nodes, wR1, (const float*)d_in[43], (const float*)d_in[44],
      (const float*)d_in[45], gsum);
  finalize_kernel<<<1, 64, 0, stream>>>(gsum, num_nodes, (const float*)d_in[46],
                                        (const float*)d_in[47], (float*)d_out);
}
